// BinaryVGG16_44384192037247
// MI455X (gfx1250) — compile-verified, matrix operand order checked by reading, not run on hardware
//
#include <hip/hip_runtime.h>
#include <stdint.h>

typedef __attribute__((ext_vector_type(8))) int v8i;
#define NTILES 4

// Permute channel index within each 64-byte K block so that each WMMA
// fragment (dword pattern {0,1,4,5,8,9,12,13}, +2 dwords for lanes 16-31)
// becomes 32 contiguous bytes: lanes 0-15 -> bytes [0,32), lanes 16-31 -> [32,64).
__device__ __forceinline__ int perm64(int c) {
    return (((c >> 3) & 1) << 5) + (((c >> 4) & 3) << 3) + (c & 7);
}

// ---------------------------------------------------------------------------
// Weight prep for binary layers l=1..12:
//   alpha[o] = min(1, mean|w|), scale[o] = alpha*g/sqrt(v+eps),
//   shift[o] = b - m*g/sqrt(v+eps),
//   wq[o][k''] = sign(w) int8, k' = (ky*3+kx)*Cin + cin, k'' = block-permuted k'
// ---------------------------------------------------------------------------
__global__ void prep_wq_kernel(const float* __restrict__ w,
                               const float* __restrict__ g, const float* __restrict__ bb,
                               const float* __restrict__ m, const float* __restrict__ v,
                               int8_t* __restrict__ wq, float* __restrict__ sc,
                               float* __restrict__ sh, int Cin) {
    const int o = blockIdx.x;
    const int K = Cin * 9;
    const float* wrow = w + (size_t)o * K;          // OIHW row for this o
    __shared__ float red[256];
    float s = 0.f;
    for (int k = threadIdx.x; k < K; k += 256) s += fabsf(wrow[k]);
    red[threadIdx.x] = s;
    __syncthreads();
    for (int d = 128; d > 0; d >>= 1) {
        if (threadIdx.x < d) red[threadIdx.x] += red[threadIdx.x + d];
        __syncthreads();
    }
    float alpha = red[0] / (float)K;
    if (alpha > 1.f) alpha = 1.f;                   // clamp(sign*alpha,-1,1)
    if (threadIdx.x == 0) {
        float scale = g[o] * rsqrtf(v[o] + 1e-5f);
        sc[o] = alpha * scale;
        sh[o] = bb[o] - m[o] * scale;
    }
    for (int k = threadIdx.x; k < K; k += 256) {
        int i = k / 9, p = k % 9;                   // src k = cin*9 + (ky*3+kx)
        float xv = wrow[k];
        int8_t q = (xv > 0.f) ? (int8_t)1 : ((xv < 0.f) ? (int8_t)-1 : (int8_t)0);
        int kp = p * Cin + i;                       // GEMM-K index
        int kpp = (kp & ~63) | perm64(kp & 63);     // block permutation
        wq[(size_t)o * K + kpp] = q;
    }
}

__global__ void prep_bn0_kernel(const float* __restrict__ g, const float* __restrict__ bb,
                                const float* __restrict__ m, const float* __restrict__ v,
                                float* __restrict__ sc, float* __restrict__ sh) {
    int o = threadIdx.x;
    if (o < 64) {
        float s = g[o] * rsqrtf(v[o] + 1e-5f);
        sc[o] = s;
        sh[o] = bb[o] - m[o] * s;
    }
}

// ---------------------------------------------------------------------------
// Zero the 1-pixel halo ring of a padded NHWC int8 buffer [16, H+2, W+2, C].
// ---------------------------------------------------------------------------
__global__ void zero_halo_kernel(int8_t* __restrict__ buf, int Hh, int Ww, int C) {
    const int Hp = Hh + 2, Wp = Ww + 2;
    const int ncells = 2 * Wp + 2 * Hh;             // border cells per image
    const int c16 = C >> 4;
    long idx = (long)blockIdx.x * blockDim.x + threadIdx.x;
    long total = (long)16 * ncells * c16;
    if (idx >= total) return;
    int cc = (int)(idx % c16);
    long q = idx / c16;
    int bc = (int)(q % ncells);
    int n = (int)(q / ncells);
    int y, x;
    if (bc < Wp)            { y = 0;      x = bc; }
    else if (bc < 2 * Wp)   { y = Hp - 1; x = bc - Wp; }
    else { int r = bc - 2 * Wp; y = 1 + (r >> 1); x = (r & 1) ? (Wp - 1) : 0; }
    int4* p = (int4*)(buf + ((((size_t)n * Hp + y) * Wp + x) * C) + (size_t)cc * 16);
    *p = make_int4(0, 0, 0, 0);
}

// ---------------------------------------------------------------------------
// Layer 0: fp32 direct conv 3->64, pad 1, + BN + binarize -> padded int8 NHWC
// (channel position permuted per perm64)
// ---------------------------------------------------------------------------
__global__ void conv0_kernel(const float* __restrict__ x, const float* __restrict__ w,
                             const float* __restrict__ sc, const float* __restrict__ sh,
                             int8_t* __restrict__ act) {
    const int o = threadIdx.x;                      // 0..63 out channel
    const int p = blockIdx.x;                       // pixel over 16*224*224
    const int xc = p % 224;
    int t = p / 224;
    const int y = t % 224;
    const int n = t / 224;
    float acc = 0.f;
#pragma unroll
    for (int c = 0; c < 3; ++c)
#pragma unroll
        for (int ky = 0; ky < 3; ++ky) {
            int yy = y + ky - 1;
            if (yy < 0 || yy >= 224) continue;
#pragma unroll
            for (int kx = 0; kx < 3; ++kx) {
                int xx = xc + kx - 1;
                if (xx < 0 || xx >= 224) continue;
                acc += x[(((size_t)n * 3 + c) * 224 + yy) * 224 + xx] *
                       w[((o * 3 + c) * 3 + ky) * 3 + kx];
            }
        }
    float yv = acc * sc[o] + sh[o];
    act[((((size_t)n * 226 + (y + 1)) * 226) + (xc + 1)) * 64 + perm64(o)] =
        (yv >= 0.f) ? (int8_t)1 : (int8_t)-1;
}

// ---------------------------------------------------------------------------
// Binary conv as implicit GEMM on V_WMMA_I32_16X16X64_IU8.
// Fully compile-time shape (CIN, DIM): constant-divide pixel decomposition,
// compile-time taps, halo-padded + block-permuted layout -> inner loop is a
// pure (s_clause b128 loads + WMMA) stream with no VALU (no WAR nop bursts).
// ---------------------------------------------------------------------------
template <int CIN, int DIM>
__global__ __launch_bounds__(128) void bingemm_kernel(
        const int8_t* __restrict__ act_in,          // padded [16,DIM+2,DIM+2,CIN]
        const int8_t* __restrict__ wq,              // [Cout][9*CIN] permuted
        const float* __restrict__ sc, const float* __restrict__ sh,
        int8_t* __restrict__ act_out,               // padded [16,DIM+2,DIM+2,Cout]
        int Cout) {
    constexpr int K = 9 * CIN;
    constexpr int CCH = CIN / 64;
    constexpr int Wp = DIM + 2;
    const int lane = threadIdx.x & 31;
    const int wave = threadIdx.x >> 5;
    const int mbase = (blockIdx.y * 4 + wave) * 16;
    const long pbase = (long)blockIdx.x * 64;
    const int col = lane & 15;
    const int kofs = (lane & 16) ? 32 : 0;          // lane-half fragment base

    long cell[NTILES];
    unsigned voff[NTILES];
#pragma unroll
    for (int t = 0; t < NTILES; ++t) {
        long p = pbase + t * 16 + col;
        int xx = (int)(p % DIM);                    // constant divides
        long q = p / DIM;
        int yy = (int)(q % DIM);
        int nn = (int)(q / DIM);
        cell[t] = ((long)nn * (DIM + 2) + (yy + 1)) * Wp + (xx + 1);
        voff[t] = (unsigned)(cell[t] * CIN) + (unsigned)kofs;
    }

    const unsigned aoff = (unsigned)(mbase + col) * (unsigned)K + (unsigned)kofs;

    v8i acc[NTILES];
#pragma unroll
    for (int t = 0; t < NTILES; ++t) acc[t] = (v8i){0, 0, 0, 0, 0, 0, 0, 0};

    for (int cb = 0; cb < CCH; ++cb) {
        const int8_t* aw = wq + (size_t)cb * 64 + aoff;     // per-lane A row base
        const int8_t* bbase = act_in + cb * 64;             // uniform
        __builtin_prefetch(aw + 64, 0, 1);                  // next channel chunk
#pragma unroll
        for (int kk = 0; kk < 9; ++kk) {
            constexpr int DY[9] = {-1, -1, -1, 0, 0, 0, 1, 1, 1};
            constexpr int DX[9] = {-1, 0, 1, -1, 0, 1, -1, 0, 1};
            const int tap = (DY[kk] * Wp + DX[kk]) * CIN;   // compile-time
            const int4* ap = (const int4*)(aw + kk * CIN);
            int4 A0 = ap[0], A1 = ap[1];                    // 2x b128
            v8i a = (v8i){A0.x, A0.y, A0.z, A0.w, A1.x, A1.y, A1.z, A1.w};
            const int8_t* bbk = bbase + tap;                // uniform saddr part
#pragma unroll
            for (int t = 0; t < NTILES; ++t) {
                const int4* bp = (const int4*)(bbk + voff[t]);
                int4 B0 = bp[0], B1 = bp[1];                // 2x b128
                v8i b = (v8i){B0.x, B0.y, B0.z, B0.w, B1.x, B1.y, B1.z, B1.w};
                acc[t] = __builtin_amdgcn_wmma_i32_16x16x64_iu8(
                    true, a, true, b, acc[t], false, false);
            }
        }
    }

    // Epilogue: i32 C/D layout -> alpha*BN -> sign -> packed b64 store
    const int orow0 = (lane & 16) ? 8 : 0;
    const int o0 = mbase + orow0;                   // multiple of 8
    float scv[8], shv[8];
#pragma unroll
    for (int r = 0; r < 8; ++r) {
        scv[r] = sc[o0 + r];
        shv[r] = sh[o0 + r];
    }
    const int o0p = (o0 & ~63) | perm64(o0 & 63);   // perm keeps +r contiguous, r<8
#pragma unroll
    for (int t = 0; t < NTILES; ++t) {
        unsigned long long pk = 0;
#pragma unroll
        for (int r = 0; r < 8; ++r) {
            float yv = (float)acc[t][r] * scv[r] + shv[r];
            unsigned char sb = (yv >= 0.f) ? (unsigned char)1 : (unsigned char)0xFF;
            pk |= (unsigned long long)sb << (8 * r);
        }
        *(unsigned long long*)(act_out + cell[t] * Cout + o0p) = pk;
    }
}

// ---------------------------------------------------------------------------
// 2x2 maxpool, padded int8 NHWC in (H+2,W+2) -> padded out (H/2+2, W/2+2)
// (channel-elementwise: permutation-invariant)
// ---------------------------------------------------------------------------
__global__ void maxpool_kernel(const int8_t* __restrict__ in, int8_t* __restrict__ out,
                               int Hh, int Ww, int C) {
    const int Ho = Hh >> 1, Wo = Ww >> 1;
    const int Wpi = Ww + 2, Wpo = Wo + 2;
    const int C4 = C >> 2;
    long idx = (long)blockIdx.x * blockDim.x + threadIdx.x;
    long total = (long)16 * Ho * Wo * C4;
    if (idx >= total) return;
    int c4 = (int)(idx % C4);
    long q = idx / C4;
    int ox = (int)(q % Wo); q /= Wo;
    int oy = (int)(q % Ho);
    int n  = (int)(q / Ho);
    const int8_t* base = in + ((((size_t)n * (Hh + 2) + (1 + 2 * oy)) * Wpi
                                + (1 + 2 * ox)) * C) + c4 * 4;
    const size_t rs = (size_t)Wpi * C;
    int8_t* ob = out + ((((size_t)n * (Ho + 2) + (1 + oy)) * Wpo + (1 + ox)) * C) + c4 * 4;
#pragma unroll
    for (int j = 0; j < 4; ++j) {
        int8_t a0 = base[j], a1 = base[C + j], a2 = base[rs + j], a3 = base[rs + C + j];
        int8_t m0 = a0 > a1 ? a0 : a1;
        int8_t m1 = a2 > a3 ? a2 : a3;
        ob[j] = m0 > m1 ? m0 : m1;
    }
}

// ---------------------------------------------------------------------------
// Global avgpool (7x7, padded 9x9 input, permuted channels) + FC 512->10
// ---------------------------------------------------------------------------
__global__ void head_kernel(const int8_t* __restrict__ act,   // padded [16,9,9,512]
                            const float* __restrict__ fcw, const float* __restrict__ fcb,
                            float* __restrict__ out) {
    const int n = blockIdx.x;
    __shared__ float avg[512];
    for (int c = threadIdx.x; c < 512; c += blockDim.x) {
        int cp = (c & ~63) | perm64(c & 63);
        int s = 0;
        for (int r = 0; r < 7; ++r)
            for (int cx = 0; cx < 7; ++cx)
                s += act[((((size_t)n * 9 + (1 + r)) * 9) + (1 + cx)) * 512 + cp];
        avg[c] = (float)s / 49.f;
    }
    __syncthreads();
    if (threadIdx.x < 10) {
        float s = fcb[threadIdx.x];
        for (int c = 0; c < 512; ++c) s += avg[c] * fcw[threadIdx.x * 512 + c];
        out[n * 10 + threadIdx.x] = s;
    }
}

// ---------------------------------------------------------------------------
// Host-side orchestration (all launches stream-ordered; graph-capture safe)
// ---------------------------------------------------------------------------
static inline void launch_bingemm(int cin, int dim, dim3 grid, hipStream_t stream,
                                  const int8_t* in, const int8_t* wq,
                                  const float* sc, const float* sh,
                                  int8_t* out, int cout) {
    const int key = cin * 1024 + dim;
    switch (key) {
        case  64 * 1024 + 224: bingemm_kernel<64, 224> <<<grid, 128, 0, stream>>>(in, wq, sc, sh, out, cout); break;
        case  64 * 1024 + 112: bingemm_kernel<64, 112> <<<grid, 128, 0, stream>>>(in, wq, sc, sh, out, cout); break;
        case 128 * 1024 + 112: bingemm_kernel<128, 112><<<grid, 128, 0, stream>>>(in, wq, sc, sh, out, cout); break;
        case 128 * 1024 +  56: bingemm_kernel<128, 56> <<<grid, 128, 0, stream>>>(in, wq, sc, sh, out, cout); break;
        case 256 * 1024 +  56: bingemm_kernel<256, 56> <<<grid, 128, 0, stream>>>(in, wq, sc, sh, out, cout); break;
        case 256 * 1024 +  28: bingemm_kernel<256, 28> <<<grid, 128, 0, stream>>>(in, wq, sc, sh, out, cout); break;
        case 512 * 1024 +  28: bingemm_kernel<512, 28> <<<grid, 128, 0, stream>>>(in, wq, sc, sh, out, cout); break;
        default:               bingemm_kernel<512, 14> <<<grid, 128, 0, stream>>>(in, wq, sc, sh, out, cout); break;
    }
}

extern "C" void kernel_launch(void* const* d_in, const int* in_sizes, int n_in,
                              void* d_out, int out_size, void* d_ws, size_t ws_size,
                              hipStream_t stream) {
    (void)in_sizes; (void)n_in; (void)out_size; (void)ws_size;
    const float* x = (const float*)d_in[0];
    const float* convw[13];
    const float *bng[13], *bnb[13], *bnm[13], *bnv[13];
    for (int i = 0; i < 13; ++i) convw[i] = (const float*)d_in[1 + i];
    for (int i = 0; i < 13; ++i) {
        bng[i] = (const float*)d_in[14 + 4 * i + 0];
        bnb[i] = (const float*)d_in[14 + 4 * i + 1];
        bnm[i] = (const float*)d_in[14 + 4 * i + 2];
        bnv[i] = (const float*)d_in[14 + 4 * i + 3];
    }
    const float* fcw = (const float*)d_in[66];
    const float* fcb = (const float*)d_in[67];

    char* ws = (char*)d_ws;
    int8_t* bufA   = (int8_t*)ws;                               // 53 MB region
    int8_t* bufB   = (int8_t*)(ws + (size_t)(53u << 20));       // 53 MB region
    int8_t* wqbase = (int8_t*)(ws + (size_t)(106u << 20));      // ~14.1 MB used
    float*  scbase = (float*) (ws + (size_t)(121u << 20));      // 13 * 1024 floats

    static const int CFGC[13] = {64, 64, 128, 128, 256, 256, 256, 512, 512, 512, 512, 512, 512};
    auto poolafter = [](int i) { return i == 1 || i == 3 || i == 6 || i == 9 || i == 12; };
    auto halo_blocks = [](int H, int W, int C) {
        long total = (long)16 * (2 * (W + 2) + 2 * H) * (C >> 4);
        return (unsigned)((total + 255) / 256);
    };

    // prep: BN0 + binarized weight/scale tables for layers 1..12
    prep_bn0_kernel<<<1, 64, 0, stream>>>(bng[0], bnb[0], bnm[0], bnv[0],
                                          scbase, scbase + 512);
    size_t wqoffs[13];
    size_t wqoff = 0;
    int cin = 64;
    for (int l = 1; l < 13; ++l) {
        int cout = CFGC[l];
        wqoffs[l] = wqoff;
        prep_wq_kernel<<<cout, 256, 0, stream>>>(convw[l], bng[l], bnb[l], bnm[l], bnv[l],
                                                 wqbase + wqoff,
                                                 scbase + (size_t)l * 1024,
                                                 scbase + (size_t)l * 1024 + 512, cin);
        wqoff += (size_t)cout * cin * 9;
        cin = cout;
    }

    // layer 0: zero halo, then fp32 conv + BN + binarize -> padded bufA
    zero_halo_kernel<<<halo_blocks(224, 224, 64), 256, 0, stream>>>(bufA, 224, 224, 64);
    conv0_kernel<<<16 * 224 * 224, 64, 0, stream>>>(x, convw[0], scbase, scbase + 512, bufA);

    int8_t* cur = bufA;
    int8_t* alt = bufB;
    int dim = 224;
    cin = 64;
    for (int l = 1; l < 13; ++l) {
        int cout = CFGC[l];
        long pixels = (long)16 * dim * dim;
        dim3 grid((unsigned)(pixels / 64), (unsigned)(cout / 64));
        zero_halo_kernel<<<halo_blocks(dim, dim, cout), 256, 0, stream>>>(alt, dim, dim, cout);
        launch_bingemm(cin, dim, grid, stream, cur, wqbase + wqoffs[l],
                       scbase + (size_t)l * 1024, scbase + (size_t)l * 1024 + 512,
                       alt, cout);
        if (poolafter(l)) {
            zero_halo_kernel<<<halo_blocks(dim / 2, dim / 2, cout), 256, 0, stream>>>(
                cur, dim / 2, dim / 2, cout);
            long tot = (long)16 * (dim / 2) * (dim / 2) * (cout / 4);
            maxpool_kernel<<<(unsigned)((tot + 255) / 256), 256, 0, stream>>>(alt, cur,
                                                                              dim, dim, cout);
            dim /= 2;                                // result stays in cur
        } else {
            int8_t* t = cur; cur = alt; alt = t;     // result in cur after swap
        }
        cin = cout;
    }

    head_kernel<<<16, 256, 0, stream>>>(cur, fcw, fcb, (float*)d_out);
}
